// DualNodeEdgeBlock_47158740910659
// MI455X (gfx1250) — matrix operand order repaired and checked by reading, not compiled, not run on hardware
//
#include <hip/hip_runtime.h>
#include <math.h>

typedef __attribute__((ext_vector_type(16))) _Float16 v16h;
typedef __attribute__((ext_vector_type(8)))  _Float16 v8h;
typedef __attribute__((ext_vector_type(8)))  float    v8f;

#define BSZ 4
#define NN  128
#define DXD 256
#define DED 64
#define DYD 64
#define NHD 8

static __device__ __forceinline__ v8f wmma16x16x32(v16h a, v16h b, v8f c) {
  return __builtin_amdgcn_wmma_f32_16x16x32_f16(false, a, false, b, (short)0, c,
                                                false, false);
}

// A fragment from row-major f16 LDS: two 16B contiguous chunks -> 2x ds_load_b128.
static __device__ __forceinline__ v16h frag_a_h(const _Float16* A, int ld, int lane) {
  int m  = lane & 15;
  int hb = (lane >> 4) & 1;
  const _Float16* p = A + (size_t)m * ld + hb * 8;
  v8h lo = *(const v8h*)p;
  v8h hi = *(const v8h*)(p + 16);
  return __builtin_shufflevector(lo, hi, 0, 1, 2, 3, 4, 5, 6, 7,
                                 8, 9, 10, 11, 12, 13, 14, 15);
}

// B fragment from prepacked fragment-major f16: one contiguous 32B load.
static __device__ __forceinline__ v16h frag_bp(const _Float16* P, int nkc, int nt,
                                               int kc, int lane) {
  return *(const v16h*)(P + (((size_t)nt * nkc + kc) * 32 + lane) * 16);
}

// ------------------------------------------------------------- pack kernel
// Convert row-major f32 W (K x N) into f16 fragment-major layout.
struct PackArgs { const float* W; _Float16* dst; int K; int N; };

__global__ void __launch_bounds__(256) pack_kernel(PackArgs a) {
  int nkc = a.K / 32;
  int total = (a.N / 16) * nkc * 32 * 16;
  for (int idx = blockIdx.x * blockDim.x + threadIdx.x; idx < total;
       idx += gridDim.x * blockDim.x) {
    int e  = idx & 15;
    int l  = (idx >> 4) & 31;
    int r  = idx >> 9;
    int kc = r % nkc;
    int nt = r / nkc;
    int n = nt * 16 + (l & 15);
    int k = kc * 32 + (e >> 3) * 16 + ((l >> 4) & 1) * 8 + (e & 7);
    a.dst[idx] = (_Float16)a.W[(size_t)k * a.N + n];
  }
}

// ---------------------------------------------------------------- kernel 1
struct NodeLinArgs {
  const float* X; const int* nmask;
  const _Float16* W[6]; const float* Bv[6];
  float* O[6];
};

__global__ void __launch_bounds__(256) node_qkv_kernel(NodeLinArgs a) {
  __shared__ _Float16 sA[16][DXD];      // f16-staged activation tile
  int tid = threadIdx.x, lane = tid & 31, wave = tid >> 5;
  int rt = blockIdx.x;          // 16-row tile of M=512
  int p  = blockIdx.y;          // which projection
  const _Float16* W = a.W[p];
  const float* bias = a.Bv[p];
  float* O = a.O[p];
  const float* Arow = a.X + (size_t)rt * 16 * DXD;
  for (int u = tid; u < 16 * DXD; u += 256)
    sA[u >> 8][u & 255] = (_Float16)Arow[u];
  __syncthreads();
#pragma unroll
  for (int s = 0; s < 2; ++s) {
    int nt = wave * 2 + s, n0 = nt * 16;
    v8f acc = {};
    for (int kc = 0; kc < DXD / 32; ++kc) {
      v16h af = frag_a_h(&sA[0][0] + kc * 32, DXD, lane);
      v16h bf = frag_bp(W, DXD / 32, nt, kc, lane);
      acc = wmma16x16x32(af, bf, acc);
    }
    int n = n0 + (lane & 15);
    int hb = (lane >> 4) & 1;
    float bv = bias[n];
#pragma unroll
    for (int r = 0; r < 8; ++r) {
      int R = rt * 16 + r + 8 * hb;
      float mk = a.nmask[R] ? 1.f : 0.f;
      O[(size_t)R * DXD + n] = (acc[r] + bv) * mk;
    }
  }
}

// ---------------------------------------------------------------- kernel 2
struct YModArgs {
  const float* y;
  const float* W[4]; const float* Bv[4];
  float* O[4];
};

__global__ void __launch_bounds__(256) ymod_kernel(YModArgs a) {
  int t = threadIdx.x;
#pragma unroll
  for (int v = 0; v < 4; ++v)
    for (int b = 0; b < BSZ; ++b) {
      float acc = a.Bv[v][t];
      for (int k = 0; k < DYD; ++k) acc += a.y[b * DYD + k] * a.W[v][k * DXD + t];
      a.O[v][b * DXD + t] = acc;
    }
}

// ---------------------------------------------------------------- kernel 3
__global__ void __launch_bounds__(256) estat_kernel(const float* E, float* sum_o,
                                                    float* sq_o, float* mn_o,
                                                    float* mx_o) {
  __shared__ float ssum[256], ssq[256], smn[256], smx[256];
  int cid = blockIdx.x;                  // b*64 + de
  int b = cid >> 6, de = cid & 63;
  int t = threadIdx.x;
  const float* base = E + (size_t)b * NN * NN * DED + de;
  float s = 0.f, q = 0.f, mn = INFINITY, mx = -INFINITY;
  for (int e = t; e < NN * NN; e += 256) {
    float v = base[(size_t)e * DED];
    s += v; q += v * v; mn = fminf(mn, v); mx = fmaxf(mx, v);
  }
  ssum[t] = s; ssq[t] = q; smn[t] = mn; smx[t] = mx;
  __syncthreads();
  for (int off = 128; off > 0; off >>= 1) {
    if (t < off) {
      ssum[t] += ssum[t + off]; ssq[t] += ssq[t + off];
      smn[t] = fminf(smn[t], smn[t + off]);
      smx[t] = fmaxf(smx[t], smx[t + off]);
    }
    __syncthreads();
  }
  if (t == 0) { sum_o[cid] = ssum[0]; sq_o[cid] = ssq[0];
                mn_o[cid] = smn[0];   mx_o[cid] = smx[0]; }
}

// ---------------------------------------------------------------- kernel 4
struct EdgeArgs {
  const float* E; const int* nmask; const float* X;
  const float* Qs; const float* Ks; const float* Vs;
  const float* Qt; const float* Kt; const float* Vt;
  const float* ye_add; const float* ye_mul;
  const float* yx_add; const float* yx_mul;
  const _Float16* pk_ems; const float* b_ems;
  const _Float16* pk_eas; const float* b_eas;
  const _Float16* pk_emt; const float* b_emt;
  const _Float16* pk_eat; const float* b_eat;
  const float* w_eg;  const float* b_eg;
  const float* w_egt; const float* b_egt;
  const _Float16* pk_eout; const float* b_eout;
  const _Float16* pk_gx;   const float* b_gx;
  const _Float16* pk_xout; const float* b_xout;
  float* outX; float* outE;
};

__global__ void __launch_bounds__(256) edge_fused_kernel(EdgeArgs a) {
  __shared__ float sE1s[16][DXD], sE2s[16][DXD], sE1t[16][DXD], sE2t[16][DXD];
  __shared__ _Float16 sEsh[16][DED], sEth[16][DED];     // f16-staged edge rows
  __shared__ _Float16 sYmodH[16][DXD];                  // f16 ymod / newX_pre
  __shared__ _Float16 sCatH[16][2 * DXD];               // f16 [X | wV]
  __shared__ float sKs[DXD], sKt[DXD], sVs[DXD], sVt[DXD];
  __shared__ float sYe1[DXD], sYe2[DXD], sYx1[DXD], sYx2[DXD];
  __shared__ float sGs[16][NHD], sGt[16][NHD];
  __shared__ float sXmI[16];

  const int tid = threadIdx.x, lane = tid & 31, wave = tid >> 5;
  const int i0 = blockIdx.x * 16;       // query tile
  const int b  = blockIdx.y;            // batch
  const int ti = tid >> 4;              // owned row slot 0..15
  const int fb = (tid & 15) * 16;       // owned contiguous f range [fb, fb+16)
  const int hh = fb >> 5;               // head index (constant per thread)

  if (tid < 16) sXmI[tid] = a.nmask[b * NN + i0 + tid] ? 1.f : 0.f;
  sYe1[tid] = a.ye_add[b * DXD + tid];
  sYe2[tid] = a.ye_mul[b * DXD + tid];
  sYx1[tid] = a.yx_add[b * DXD + tid];
  sYx2[tid] = a.yx_mul[b * DXD + tid];

  float qs[16], qt[16], am[16], al[16], av[16];
  {
    const float* qsrow = a.Qs + (size_t)(b * NN + i0 + ti) * DXD + fb;
    const float* qtrow = a.Qt + (size_t)(b * NN + i0 + ti) * DXD + fb;
#pragma unroll
    for (int r = 0; r < 16; ++r) {
      qs[r] = qsrow[r];
      qt[r] = qtrow[r];
      am[r] = -INFINITY; al[r] = 0.f; av[r] = 0.f;
    }
  }
  __syncthreads();

  const float scale = 0.1767766952966369f;   // 1/sqrt(DF=32)

  for (int jm = 0; jm < NN; ++jm) {
    // ---- stage edge rows (both orientations, f16) + key/value rows ----
    for (int u = tid; u < 16 * DED; u += 256) {
      int i = u >> 6, c = u & 63;
      sEsh[i][c] = (_Float16)a.E[((size_t)(b * NN + i0 + i) * NN + jm) * DED + c];
      sEth[i][c] = (_Float16)a.E[((size_t)(b * NN + jm) * NN + (i0 + i)) * DED + c];
    }
    {
      size_t jrow = (size_t)(b * NN + jm) * DXD + tid;
      sKs[tid] = a.Ks[jrow]; sKt[tid] = a.Kt[jrow];
      sVs[tid] = a.Vs[jrow]; sVt[tid] = a.Vt[jrow];
    }
    if (jm + 1 < NN)
      __builtin_prefetch(&a.E[((size_t)(b * NN + i0 + ti) * NN + jm + 1) * DED], 0, 1);
    __syncthreads();
    const float xmj = a.nmask[b * NN + jm] ? 1.f : 0.f;

    // ---- four K=64 -> 256 edge projections via WMMA ----
    {
      int p = wave >> 1;                 // projection handled by this wave pair
      int ntb = (wave & 1) * 8;          // 8 N-tiles per wave
      const _Float16* src = (p < 2) ? &sEsh[0][0] : &sEth[0][0];
      const _Float16* Wp; const float* bias; float* dst;
      switch (p) {
        case 0:  Wp = a.pk_ems; bias = a.b_ems; dst = &sE1s[0][0]; break;
        case 1:  Wp = a.pk_eas; bias = a.b_eas; dst = &sE2s[0][0]; break;
        case 2:  Wp = a.pk_emt; bias = a.b_emt; dst = &sE1t[0][0]; break;
        default: Wp = a.pk_eat; bias = a.b_eat; dst = &sE2t[0][0]; break;
      }
      // A fragments are identical for all 8 tiles: hoist.
      v16h af0 = frag_a_h(src, DED, lane);
      v16h af1 = frag_a_h(src + 32, DED, lane);
      int hb = (lane >> 4) & 1;
#pragma unroll
      for (int q = 0; q < 8; ++q) {
        int nt = ntb + q, n0 = nt * 16;
        v8f acc = {};
        acc = wmma16x16x32(af0, frag_bp(Wp, 2, nt, 0, lane), acc);
        acc = wmma16x16x32(af1, frag_bp(Wp, 2, nt, 1, lane), acc);
        int n = n0 + (lane & 15);
        float bv = bias[n];
#pragma unroll
        for (int r = 0; r < 8; ++r) {
          int m = r + 8 * hb;
          dst[m * DXD + n] = (acc[r] + bv) * sXmI[m] * xmj;
        }
      }
    }
    // ---- edge gates (tiny DE->NH dots, VALU) ----
    {
      int isT = tid >= 128, v = tid & 127;
      int i = v >> 3, h = v & 7;
      const _Float16* src = isT ? &sEth[i][0] : &sEsh[i][0];
      const float* W = isT ? a.w_egt : a.w_eg;
      float acc = (isT ? a.b_egt : a.b_eg)[h];
      for (int c = 0; c < DED; ++c) acc += (float)src[c] * W[c * NHD + h];
      float g = 1.f / (1.f + __expf(-acc));
      if (isT) sGt[i][h] = g; else sGs[i][h] = g;
    }
    __syncthreads();

    // ---- elementwise Y_st/Y_ts + online-softmax update + ye modulation ----
    {
      bool kvalid = xmj != 0.f;
      float gs_ = sGs[ti][hh], gt_ = sGt[ti][hh];
#pragma unroll
      for (int r = 0; r < 16; ++r) {
        int f = fb + r;
        float e1 = sE1s[ti][f], e2 = sE2s[ti][f];
        float y1 = (qs[r] * sKt[f] * scale) * (e1 + 1.f) + e2;
        float yst = y1 * (gs_ * (e1 + 1.f)) + e2;
        float e1t = sE1t[ti][f], e2t = sE2t[ti][f];
        float y2 = (qt[r] * sKs[f] * scale) * (e1t + 1.f) + e2t;
        float yts = y2 * (gt_ * (e1t + 1.f)) + e2t;
        if (kvalid) {
          float mn = fmaxf(am[r], yst);
          float c0 = __expf(am[r] - mn), p0 = __expf(yst - mn);
          al[r] = al[r] * c0 + p0; av[r] = av[r] * c0 + p0 * sVt[f]; am[r] = mn;
          mn = fmaxf(am[r], yts);
          c0 = __expf(am[r] - mn); p0 = __expf(yts - mn);
          al[r] = al[r] * c0 + p0; av[r] = av[r] * c0 + p0 * sVs[f]; am[r] = mn;
        }
        sYmodH[ti][f] = (_Float16)(sYe1[f] + (sYe2[f] + 1.f) * yst);
      }
    }
    __syncthreads();

    // ---- e_out: 256 -> 64 projection, residual+mask, store newE ----
    if (wave < 4) {
      int n0 = wave * 16;
      v8f acc = {};
#pragma unroll
      for (int kc = 0; kc < 8; ++kc) {
        v16h af = frag_a_h(&sYmodH[0][0] + kc * 32, DXD, lane);
        v16h bf = frag_bp(a.pk_eout, 8, wave, kc, lane);
        acc = wmma16x16x32(af, bf, acc);
      }
      int n = n0 + (lane & 15), hb = (lane >> 4) & 1;
      float bv = a.b_eout[n];
#pragma unroll
      for (int r = 0; r < 8; ++r) {
        int m = r + 8 * hb;
        float em = sXmI[m] * xmj;
        size_t off = ((size_t)(b * NN + i0 + m) * NN + jm) * DED + n;
        a.outE[off] = em * (a.E[off] + (acc[r] + bv));
      }
    }
    __syncthreads();
  }

  // ---- finalize attention; gate_x; x_out ----
  {
    const float* xrow = a.X + (size_t)(b * NN + i0 + ti) * DXD + fb;
#pragma unroll
    for (int r = 0; r < 16; ++r) {
      int f = fb + r;
      float wv = av[r] / al[r];            // weighted_V
      float xv = xrow[r];                  // X residual
      sE1s[ti][f] = wv;
      sE1t[ti][f] = xv;
      sCatH[ti][f] = (_Float16)xv;         // [X | wV] f16 for gate_x A-side
      sCatH[ti][DXD + f] = (_Float16)wv;
    }
  }
  __syncthreads();

#pragma unroll
  for (int s = 0; s < 2; ++s) {
    int nt = wave * 2 + s, n0 = nt * 16;
    v8f acc = {};
    for (int kc = 0; kc < 16; ++kc) {      // K = 512 = [X | wV]
      v16h af = frag_a_h(&sCatH[0][0] + kc * 32, 2 * DXD, lane);
      v16h bf = frag_bp(a.pk_gx, 16, nt, kc, lane);
      acc = wmma16x16x32(af, bf, acc);
    }
    int n = n0 + (lane & 15), hb = (lane >> 4) & 1;
    float bv = a.b_gx[n];
#pragma unroll
    for (int r = 0; r < 8; ++r) {
      int m = r + 8 * hb;
      float g = 1.f / (1.f + __expf(-(acc[r] + bv)));
      float xv = sE1t[m][n], wv = sE1s[m][n];
      float mod = sYx1[n] + (sYx2[n] + 1.f) * wv;
      sYmodH[m][n] = (_Float16)(g * xv + (1.f - g) * mod);      // pre x_out
    }
  }
  __syncthreads();

#pragma unroll
  for (int s = 0; s < 2; ++s) {
    int nt = wave * 2 + s, n0 = nt * 16;
    v8f acc = {};
    for (int kc = 0; kc < 8; ++kc) {
      v16h af = frag_a_h(&sYmodH[0][0] + kc * 32, DXD, lane);
      v16h bf = frag_bp(a.pk_xout, 8, nt, kc, lane);
      acc = wmma16x16x32(af, bf, acc);
    }
    int n = n0 + (lane & 15), hb = (lane >> 4) & 1;
    float bv = a.b_xout[n];
#pragma unroll
    for (int r = 0; r < 8; ++r) {
      int m = r + 8 * hb;
      a.outX[(size_t)(b * NN + i0 + m) * DXD + n] = (acc[r] + bv) * sXmI[m];
    }
  }
}

// ---------------------------------------------------------------- kernel 5
struct YFinalArgs {
  const float* X; const float* y;
  const float* est_sum; const float* est_sq;
  const float* est_mn;  const float* est_mx;
  const float* w_yy; const float* b_yy;
  const float* w_xy; const float* b_xy;
  const float* w_ey; const float* b_ey;
  const float* w_y1; const float* b_y1;
  const float* w_y2; const float* b_y2;
  float* out;
};

__global__ void __launch_bounds__(256) yfinal_kernel(YFinalArgs a) {
  __shared__ float sxs[BSZ][4 * DXD];
  __shared__ float ses[BSZ][4 * DED];
  __shared__ float sy[BSZ * DYD], st[BSZ * DYD], sh[BSZ * DYD];
  int t = threadIdx.x;
  sy[t] = a.y[t];
  for (int g = 0; g < 4; ++g) {
    int col = t + g * 256;
    int b = col >> 8, dx = col & 255;
    const float* base = a.X + (size_t)b * NN * DXD + dx;
    float s = 0.f, q = 0.f, mn = INFINITY, mx = -INFINITY;
    for (int n = 0; n < NN; ++n) {
      float v = base[(size_t)n * DXD];
      s += v; q += v * v; mn = fminf(mn, v); mx = fmaxf(mx, v);
    }
    float mean = s / NN;
    float var = (q - s * mean) / (NN - 1);
    sxs[b][0 * DXD + dx] = mean; sxs[b][1 * DXD + dx] = mn;
    sxs[b][2 * DXD + dx] = mx;   sxs[b][3 * DXD + dx] = sqrtf(fmaxf(var, 0.f));
  }
  {
    int b = t >> 6, de = t & 63;
    const float M = (float)(NN * NN);
    float s = a.est_sum[t], q = a.est_sq[t];
    float mean = s / M;
    float var = (q - s * mean) / (M - 1.f);
    ses[b][0 * DED + de] = mean;         ses[b][1 * DED + de] = a.est_mn[t];
    ses[b][2 * DED + de] = a.est_mx[t];  ses[b][3 * DED + de] = sqrtf(fmaxf(var, 0.f));
  }
  __syncthreads();
  int b = t >> 6, o = t & 63;
  {
    float acc = a.b_yy[o];
    for (int k = 0; k < DYD; ++k) acc += sy[b * DYD + k] * a.w_yy[k * DYD + o];
    float acc2 = a.b_xy[o];
    for (int k = 0; k < 4 * DXD; ++k) acc2 += sxs[b][k] * a.w_xy[k * DYD + o];
    float acc3 = a.b_ey[o];
    for (int k = 0; k < 4 * DED; ++k) acc3 += ses[b][k] * a.w_ey[k * DYD + o];
    st[t] = acc + acc2 + acc3;
  }
  __syncthreads();
  {
    float acc = a.b_y1[o];
    for (int k = 0; k < DYD; ++k) acc += st[b * DYD + k] * a.w_y1[k * DYD + o];
    sh[t] = fmaxf(acc, 0.f);
  }
  __syncthreads();
  {
    float acc = a.b_y2[o];
    for (int k = 0; k < DYD; ++k) acc += sh[b * DYD + k] * a.w_y2[k * DYD + o];
    a.out[t] = sy[t] + acc;
  }
}

// ---------------------------------------------------------------- launch
extern "C" void kernel_launch(void* const* d_in, const int* in_sizes, int n_in,
                              void* d_out, int out_size, void* d_ws, size_t ws_size,
                              hipStream_t stream) {
  (void)in_sizes; (void)n_in; (void)out_size; (void)ws_size;
  const float* X = (const float*)d_in[0];
  const float* E = (const float*)d_in[1];
  const float* y = (const float*)d_in[2];
  const int*   nmask = (const int*)d_in[3];

  enum { P_q_s, P_k_s, P_v_s, P_q_t, P_k_t, P_v_t,
         P_y_e_mul, P_y_e_add, P_y_x_mul, P_y_x_add,
         P_e_mul_s, P_e_add_s, P_e_mul_t, P_e_add_t,
         P_y_y, P_x_y, P_e_y, P_x_out, P_e_out,
         P_y_out1, P_y_out2, P_gate_x, P_edge_gate, P_edge_gate_t };
  auto W = [&](int i) { return (const float*)d_in[4 + 2 * i]; };
  auto B = [&](int i) { return (const float*)d_in[5 + 2 * i]; };

  float* ws = (float*)d_ws;
  float* Qs = ws;           float* Ks = ws + 131072;  float* Vs = ws + 262144;
  float* Qt = ws + 393216;  float* Kt = ws + 524288;  float* Vt = ws + 655360;
  float* YEM = ws + 786432; float* YEA = YEM + 1024;
  float* YXM = YEA + 1024;  float* YXA = YXM + 1024;
  float* ESUM = YXA + 1024; float* ESQ = ESUM + 256;
  float* EMN = ESQ + 256;   float* EMX = EMN + 256;

  // packed f16 weights after the f32 region (791552 floats used)
  _Float16* pk = (_Float16*)(ws + 791552);
  _Float16* PK_QKV[6];
  for (int i = 0; i < 6; ++i) PK_QKV[i] = pk + (size_t)i * 65536;
  _Float16* PK_EMS = pk + 393216;
  _Float16* PK_EAS = PK_EMS + 16384;
  _Float16* PK_EMT = PK_EAS + 16384;
  _Float16* PK_EAT = PK_EMT + 16384;
  _Float16* PK_EOUT = PK_EAT + 16384;
  _Float16* PK_GX   = PK_EOUT + 16384;
  _Float16* PK_XOUT = PK_GX + 131072;

  float* outX = (float*)d_out;
  float* outE = outX + BSZ * NN * DXD;
  float* outY = outE + (size_t)BSZ * NN * NN * DED;

  // ---- prepack all GEMM weights to f16 fragment-major ----
  {
    int qkv[6] = { P_q_s, P_k_s, P_v_s, P_q_t, P_k_t, P_v_t };
    for (int i = 0; i < 6; ++i) {
      PackArgs pa{ W(qkv[i]), PK_QKV[i], DXD, DXD };
      pack_kernel<<<64, 256, 0, stream>>>(pa);
    }
    PackArgs p1{ W(P_e_mul_s), PK_EMS, DED, DXD };  pack_kernel<<<64, 256, 0, stream>>>(p1);
    PackArgs p2{ W(P_e_add_s), PK_EAS, DED, DXD };  pack_kernel<<<64, 256, 0, stream>>>(p2);
    PackArgs p3{ W(P_e_mul_t), PK_EMT, DED, DXD };  pack_kernel<<<64, 256, 0, stream>>>(p3);
    PackArgs p4{ W(P_e_add_t), PK_EAT, DED, DXD };  pack_kernel<<<64, 256, 0, stream>>>(p4);
    PackArgs p5{ W(P_e_out),  PK_EOUT, DXD, DED };  pack_kernel<<<64, 256, 0, stream>>>(p5);
    PackArgs p6{ W(P_gate_x), PK_GX, 2 * DXD, DXD };pack_kernel<<<64, 256, 0, stream>>>(p6);
    PackArgs p7{ W(P_x_out),  PK_XOUT, DXD, DXD };  pack_kernel<<<64, 256, 0, stream>>>(p7);
  }

  NodeLinArgs na;
  na.X = X; na.nmask = nmask;
  {
    int qkv[6] = { P_q_s, P_k_s, P_v_s, P_q_t, P_k_t, P_v_t };
    float* outs[6] = { Qs, Ks, Vs, Qt, Kt, Vt };
    for (int i = 0; i < 6; ++i) { na.W[i] = PK_QKV[i]; na.Bv[i] = B(qkv[i]); na.O[i] = outs[i]; }
  }
  node_qkv_kernel<<<dim3(32, 6), 256, 0, stream>>>(na);

  YModArgs ya;
  ya.y = y;
  {
    int yv[4] = { P_y_e_mul, P_y_e_add, P_y_x_mul, P_y_x_add };
    float* youts[4] = { YEM, YEA, YXM, YXA };
    for (int i = 0; i < 4; ++i) { ya.W[i] = W(yv[i]); ya.Bv[i] = B(yv[i]); ya.O[i] = youts[i]; }
  }
  ymod_kernel<<<1, 256, 0, stream>>>(ya);

  estat_kernel<<<256, 256, 0, stream>>>(E, ESUM, ESQ, EMN, EMX);

  EdgeArgs ea;
  ea.E = E; ea.nmask = nmask; ea.X = X;
  ea.Qs = Qs; ea.Ks = Ks; ea.Vs = Vs; ea.Qt = Qt; ea.Kt = Kt; ea.Vt = Vt;
  ea.ye_add = YEA; ea.ye_mul = YEM; ea.yx_add = YXA; ea.yx_mul = YXM;
  ea.pk_ems = PK_EMS; ea.b_ems = B(P_e_mul_s);
  ea.pk_eas = PK_EAS; ea.b_eas = B(P_e_add_s);
  ea.pk_emt = PK_EMT; ea.b_emt = B(P_e_mul_t);
  ea.pk_eat = PK_EAT; ea.b_eat = B(P_e_add_t);
  ea.w_eg  = W(P_edge_gate);   ea.b_eg  = B(P_edge_gate);
  ea.w_egt = W(P_edge_gate_t); ea.b_egt = B(P_edge_gate_t);
  ea.pk_eout = PK_EOUT; ea.b_eout = B(P_e_out);
  ea.pk_gx   = PK_GX;   ea.b_gx   = B(P_gate_x);
  ea.pk_xout = PK_XOUT; ea.b_xout = B(P_x_out);
  ea.outX = outX; ea.outE = outE;
  edge_fused_kernel<<<dim3(8, 4), 256, 0, stream>>>(ea);

  YFinalArgs fa;
  fa.X = X; fa.y = y;
  fa.est_sum = ESUM; fa.est_sq = ESQ; fa.est_mn = EMN; fa.est_mx = EMX;
  fa.w_yy = W(P_y_y);    fa.b_yy = B(P_y_y);
  fa.w_xy = W(P_x_y);    fa.b_xy = B(P_x_y);
  fa.w_ey = W(P_e_y);    fa.b_ey = B(P_e_y);
  fa.w_y1 = W(P_y_out1); fa.b_y1 = B(P_y_out1);
  fa.w_y2 = W(P_y_out2); fa.b_y2 = B(P_y_out2);
  fa.out = outY;
  yfinal_kernel<<<1, 256, 0, stream>>>(fa);
}
